// TriangleMultiplicativeModuleOriginal_28372553957426
// MI455X (gfx1250) — compile-verified, matrix operand order checked by reading, not run on hardware
//
#include <hip/hip_runtime.h>
#include <hip/hip_bf16.h>

// ---------------------------------------------------------------------------
// Triangle multiplicative module for MI455X (gfx1250, wave32, WMMA).
// Pipeline: LN -> fused 5-proj GEMM (bf16 WMMA) -> per-channel triangle GEMM
// (bf16 WMMA, async global->LDS double-buffered operands) -> LN+gate+out GEMM.
// ---------------------------------------------------------------------------

typedef __bf16 bf16_t;
typedef __attribute__((ext_vector_type(16))) __bf16 v16bf;
typedef __attribute__((ext_vector_type(8)))  float  v8f;

#define NTOK 512
#define D128 128
#define MROWS (NTOK * NTOK)  // 262144

__device__ __forceinline__ float sigmoidf_(float v) {
    return 1.0f / (1.0f + __expf(-v));
}

// Load one 16x32 bf16 WMMA A/B fragment from a row-major [rows][ldk] matrix
// at (row0, k0).  Per the CDNA5 16-bit A-matrix layout, lane L (m = L&15,
// h = L>>4) holds K = h*8..h*8+7 in elems 0..7 and K = 16+h*8.. in elems 8..15
// => two 16-byte loads per lane.  Works for generic (global or LDS) pointers.
__device__ __forceinline__ v16bf frag_load(const bf16_t* __restrict__ base,
                                           size_t row0, int ldk, int k0, int lane) {
    int m = lane & 15, h = lane >> 4;
    const bf16_t* p = base + (row0 + (size_t)m) * (size_t)ldk + k0 + h * 8;
    union { v16bf v; uint4 u[2]; } t;
    t.u[0] = *(const uint4*)(p);
    t.u[1] = *(const uint4*)(p + 16);
    return t.v;
}

__device__ __forceinline__ v8f wmma_bf16(v16bf a, v16bf b, v8f c) {
    return __builtin_amdgcn_wmma_f32_16x16x32_bf16(false, a, false, b,
                                                   (short)0, c, false, false);
}

// 16x16 output tile, K=128 (4 WMMA steps) against a transposed weight [n][k].
__device__ __forceinline__ v8f gemm_tile128(const v16bf a[4],
                                            const bf16_t* __restrict__ Wt,
                                            int n0, int lane, v8f acc) {
#pragma unroll
    for (int kk = 0; kk < 4; ++kk) {
        v16bf b = frag_load(Wt, (size_t)n0, D128, kk * 32, lane);
        acc = wmma_bf16(a[kk], b, acc);
    }
    return acc;
}

// One per-lane 16-byte async DMA: global -> LDS (ASYNCcnt-tracked, no VGPRs).
__device__ __forceinline__ void async_copy16(uint32_t lds_addr, const void* gptr) {
    uint64_t ga = (uint64_t)(uintptr_t)gptr;
    asm volatile("global_load_async_to_lds_b128 %0, %1, off"
                 :: "v"(lds_addr), "v"(ga) : "memory");
}

__device__ __forceinline__ void wait_async0() {
    asm volatile("s_wait_asynccnt 0x0" ::: "memory");
}

// ---------------------------------------------------------------------------
// K1: LayerNorm over D=128, fp32 in -> bf16 out.  One row per wave (wave32:
// each lane owns 4 channels, shuffle-reduce across 32 lanes).
// ---------------------------------------------------------------------------
__global__ void k_layernorm(const float* __restrict__ x,
                            const float* __restrict__ sc,
                            const float* __restrict__ bi,
                            bf16_t* __restrict__ xn) {
    int tid = threadIdx.x, wave = tid >> 5, lane = tid & 31;
    size_t r = (size_t)blockIdx.x * 8 + wave;
    float4 xv = *(const float4*)(x + r * D128 + lane * 4);

    float s = xv.x + xv.y + xv.z + xv.w;
#pragma unroll
    for (int o = 16; o > 0; o >>= 1) s += __shfl_xor(s, o);
    float mu = s * (1.0f / D128);

    float d0 = xv.x - mu, d1 = xv.y - mu, d2 = xv.z - mu, d3 = xv.w - mu;
    float q = d0 * d0 + d1 * d1 + d2 * d2 + d3 * d3;
#pragma unroll
    for (int o = 16; o > 0; o >>= 1) q += __shfl_xor(q, o);
    float inv = rsqrtf(q * (1.0f / D128) + 1e-6f);

    float4 scv = *(const float4*)(sc + lane * 4);
    float4 bv  = *(const float4*)(bi + lane * 4);
    union { ushort4 u; __bf16 h[4]; } o;
    o.h[0] = (__bf16)(d0 * inv * scv.x + bv.x);
    o.h[1] = (__bf16)(d1 * inv * scv.y + bv.y);
    o.h[2] = (__bf16)(d2 * inv * scv.z + bv.z);
    o.h[3] = (__bf16)(d3 * inv * scv.w + bv.w);
    *(ushort4*)(xn + r * D128 + lane * 4) = o.u;
}

// ---------------------------------------------------------------------------
// K2: fused 5-projection GEMM.  All 5 weights (bf16, transposed [n][k]) live
// in 160 KB of LDS (CDNA5: 320 KB/WGP).  Each wave owns a 16-row M-tile with
// its 4 A fragments (K=128) resident, loops the 8 N-tiles computing
//   left  = (xn@Wl + bl) * mask * sigmoid(xn@Wlg + blg)
//   right = (xn@Wr + br) * mask * sigmoid(xn@Wrg + brg)
//   ogate = sigmoid(xn@Wog + bog)
// and scatters left/right into transposed triangle-GEMM operand layouts
//   A[d][i][k] (i = r%512, k = r/512)  and  Bt[d][j][k].
// ---------------------------------------------------------------------------
__global__ void k_proj(const bf16_t* __restrict__ xn,
                       const float* __restrict__ src_mask,
                       const float* __restrict__ Wl,  const float* __restrict__ bl,
                       const float* __restrict__ Wr,  const float* __restrict__ br,
                       const float* __restrict__ Wlg, const float* __restrict__ blg,
                       const float* __restrict__ Wrg, const float* __restrict__ brg,
                       const float* __restrict__ Wog, const float* __restrict__ bog,
                       bf16_t* __restrict__ Ag, bf16_t* __restrict__ Bg,
                       bf16_t* __restrict__ gate) {
    extern __shared__ __align__(16) char smem[];
    bf16_t* Wt = (bf16_t*)smem;                       // 5 * 128*128 bf16
    float*  bs = (float*)(Wt + 5 * D128 * D128);      // 5 * 128 f32

    const float* Wsrc[5] = {Wl, Wr, Wlg, Wrg, Wog};
    const float* bsrc[5] = {bl, br, blg, brg, bog};
    int tid = threadIdx.x;
#pragma unroll
    for (int w5 = 0; w5 < 5; ++w5) {
        const float* W = Wsrc[w5];
        bf16_t* dst = Wt + w5 * D128 * D128;
        for (int idx = tid; idx < D128 * D128; idx += 256) {
            int k = idx >> 7, n = idx & 127;
            dst[n * D128 + k] = (bf16_t)W[idx];       // transpose to [n][k]
        }
        if (tid < D128) bs[w5 * D128 + tid] = bsrc[w5][tid];
    }
    __syncthreads();

    int wave = tid >> 5, lane = tid & 31;
    int hl = lane >> 4, nl = lane & 15;
    size_t r0 = (size_t)blockIdx.x * 128 + wave * 16;

    v16bf a[4];
#pragma unroll
    for (int kk = 0; kk < 4; ++kk) a[kk] = frag_load(xn, r0, D128, kk * 32, lane);

    float maskv[8];
#pragma unroll
    for (int v = 0; v < 8; ++v) {
        size_t r = r0 + v + 8 * hl;
        maskv[v] = src_mask[r >> 9] * src_mask[r & 511];
    }

    for (int nt = 0; nt < 8; ++nt) {
        int n0 = nt * 16;
        int d  = n0 + nl;

        // ---- left * sigmoid(lgate), scatter to A[d][i][k]
        v8f accL = {}; accL = gemm_tile128(a, Wt + 0 * D128 * D128, n0, lane, accL);
        v8f accG = {}; accG = gemm_tile128(a, Wt + 2 * D128 * D128, n0, lane, accG);
        float bL = bs[0 * D128 + d], bG = bs[2 * D128 + d];
#pragma unroll
        for (int v = 0; v < 8; ++v) {
            size_t r = r0 + v + 8 * hl;
            int i = (int)(r & 511), k = (int)(r >> 9);
            float val = (accL[v] + bL) * maskv[v] * sigmoidf_(accG[v] + bG);
            Ag[(size_t)d * MROWS + (size_t)i * NTOK + k] = (bf16_t)val;
        }

        // ---- right * sigmoid(rgate), scatter to Bt[d][j][k]
        v8f accR = {};  accR  = gemm_tile128(a, Wt + 1 * D128 * D128, n0, lane, accR);
        v8f accRG = {}; accRG = gemm_tile128(a, Wt + 3 * D128 * D128, n0, lane, accRG);
        float bR = bs[1 * D128 + d], bRG = bs[3 * D128 + d];
#pragma unroll
        for (int v = 0; v < 8; ++v) {
            size_t r = r0 + v + 8 * hl;
            int j = (int)(r & 511), k = (int)(r >> 9);
            float val = (accR[v] + bR) * maskv[v] * sigmoidf_(accRG[v] + bRG);
            Bg[(size_t)d * MROWS + (size_t)j * NTOK + k] = (bf16_t)val;
        }

        // ---- out gate, natural [r][d] layout
        v8f accO = {}; accO = gemm_tile128(a, Wt + 4 * D128 * D128, n0, lane, accO);
        float bO = bs[4 * D128 + d];
#pragma unroll
        for (int v = 0; v < 8; ++v) {
            size_t r = r0 + v + 8 * hl;
            gate[r * D128 + d] = (bf16_t)sigmoidf_(accO[v] + bO);
        }
    }
}

// ---------------------------------------------------------------------------
// K3: triangle einsum as 128 per-channel 512x512x512 bf16 GEMMs.
// Grid (4,4,128); block = 8 waves computes a 128x128 output tile.  A/B
// 128x32 K-chunks are streamed global->LDS with GLOBAL_LOAD_ASYNC_TO_LDS_B128
// (double-buffered, ASYNCcnt + barrier sync, DMA of chunk k+1 overlapped with
// the 8 WMMAs/wave of chunk k).  LDS row stride padded 32->40 bf16 so the
// 16-lane b128 fragment reads land on distinct bank quads.
// ---------------------------------------------------------------------------
#define K3_LDK  40
#define K3_BUFE (128 * K3_LDK)   // elements per A or B buffer

__global__ void __launch_bounds__(256) k_triangle(const bf16_t* __restrict__ Ag,
                                                  const bf16_t* __restrict__ Bg,
                                                  float* __restrict__ out) {
    extern __shared__ __align__(16) char smem[];
    bf16_t* base = (bf16_t*)smem;     // layout: A0 | B0 | A1 | B1

    int d = blockIdx.z;
    const bf16_t* A = Ag + (size_t)d * MROWS;
    const bf16_t* B = Bg + (size_t)d * MROWS;
    int i0 = blockIdx.x * 128, j0 = blockIdx.y * 128;
    int tid = threadIdx.x, wave = tid >> 5, lane = tid & 31;

    // Cooperative DMA mapping: tile = 128 rows x 64 B; chunk idx -> (row, quad).
    uint32_t lds0 = (uint32_t)(uintptr_t)base;    // DS offset of smem base
    int rowA = tid >> 2, rowB = (tid + 256) >> 2, q = tid & 3;

    auto issue = [&](int kt, int b) {
        int kbase = kt * 32;
        uint32_t aoff = lds0 + (uint32_t)((2 * b) * K3_BUFE * 2);
        uint32_t boff = aoff + (uint32_t)(K3_BUFE * 2);
        async_copy16(aoff + (uint32_t)((rowA * K3_LDK + q * 8) * 2),
                     A + (size_t)(i0 + rowA) * NTOK + kbase + q * 8);
        async_copy16(aoff + (uint32_t)((rowB * K3_LDK + q * 8) * 2),
                     A + (size_t)(i0 + rowB) * NTOK + kbase + q * 8);
        async_copy16(boff + (uint32_t)((rowA * K3_LDK + q * 8) * 2),
                     B + (size_t)(j0 + rowA) * NTOK + kbase + q * 8);
        async_copy16(boff + (uint32_t)((rowB * K3_LDK + q * 8) * 2),
                     B + (size_t)(j0 + rowB) * NTOK + kbase + q * 8);
    };

    v8f zero = {};
    v8f acc[8] = {zero, zero, zero, zero, zero, zero, zero, zero};

    const int NKT = NTOK / 32;    // 16 K-chunks
    issue(0, 0);
    for (int kt = 0; kt < NKT; ++kt) {
        int cur = kt & 1;
        wait_async0();            // my chunk-kt DMAs landed in LDS
        __syncthreads();          // everyone's did
        if (kt + 1 < NKT) issue(kt + 1, cur ^ 1);   // overlap DMA with math

        const bf16_t* Ac = base + (2 * cur) * K3_BUFE;
        const bf16_t* Bc = Ac + K3_BUFE;
        v16bf af = frag_load(Ac, (size_t)(wave * 16), K3_LDK, 0, lane);
#pragma unroll
        for (int t = 0; t < 8; ++t) {
            v16bf bf = frag_load(Bc, (size_t)(t * 16), K3_LDK, 0, lane);
            acc[t] = wmma_bf16(af, bf, acc[t]);
        }
        __syncthreads();          // done reading buf[cur] before it's refilled
    }

    int hl = lane >> 4, nl = lane & 15;
#pragma unroll
    for (int t = 0; t < 8; ++t)
#pragma unroll
        for (int v = 0; v < 8; ++v) {
            int i = i0 + wave * 16 + v + 8 * hl;
            int j = j0 + t * 16 + nl;
            out[((size_t)i * NTOK + j) * D128 + d] = acc[t][v];
        }
}

// ---------------------------------------------------------------------------
// K4: LayerNorm(out) * out_gate, then @ W_out + b_out.  Each wave normalizes
// its 16 rows into a private bf16 LDS tile, then runs the 128x128 GEMM with
// W_out (transposed, bf16) in LDS.
// ---------------------------------------------------------------------------
__global__ void k_outproj(const float* __restrict__ tri,
                          const bf16_t* __restrict__ gate,
                          const float* __restrict__ ons,
                          const float* __restrict__ onb,
                          const float* __restrict__ Wout,
                          const float* __restrict__ bout,
                          float* __restrict__ y) {
    extern __shared__ __align__(16) char smem[];
    bf16_t* Wt    = (bf16_t*)smem;                    // 128*128 bf16 (transposed)
    float*  bso   = (float*)(Wt + D128 * D128);       // 128 f32
    bf16_t* tiles = (bf16_t*)(bso + D128);            // 8 waves * 16 * 128 bf16

    int tid = threadIdx.x;
    for (int idx = tid; idx < D128 * D128; idx += 256) {
        int k = idx >> 7, n = idx & 127;
        Wt[n * D128 + k] = (bf16_t)Wout[idx];
    }
    if (tid < D128) bso[tid] = bout[tid];
    __syncthreads();

    int wave = tid >> 5, lane = tid & 31;
    int hl = lane >> 4, nl = lane & 15;
    size_t r0 = (size_t)blockIdx.x * 128 + wave * 16;
    bf16_t* mytile = tiles + wave * 16 * D128;

    float4 scv = *(const float4*)(ons + lane * 4);
    float4 bbv = *(const float4*)(onb + lane * 4);

    for (int rr = 0; rr < 16; ++rr) {
        size_t r = r0 + rr;
        float4 xv = *(const float4*)(tri + r * D128 + lane * 4);
        float s = xv.x + xv.y + xv.z + xv.w;
#pragma unroll
        for (int o = 16; o > 0; o >>= 1) s += __shfl_xor(s, o);
        float mu = s * (1.0f / D128);
        float d0 = xv.x - mu, d1 = xv.y - mu, d2 = xv.z - mu, d3 = xv.w - mu;
        float q = d0 * d0 + d1 * d1 + d2 * d2 + d3 * d3;
#pragma unroll
        for (int o = 16; o > 0; o >>= 1) q += __shfl_xor(q, o);
        float inv = rsqrtf(q * (1.0f / D128) + 1e-6f);

        union { ushort4 u; __bf16 h[4]; } g;
        g.u = *(const ushort4*)(gate + r * D128 + lane * 4);

        union { ushort4 u; __bf16 h[4]; } o;
        o.h[0] = (__bf16)((d0 * inv * scv.x + bbv.x) * (float)g.h[0]);
        o.h[1] = (__bf16)((d1 * inv * scv.y + bbv.y) * (float)g.h[1]);
        o.h[2] = (__bf16)((d2 * inv * scv.z + bbv.z) * (float)g.h[2]);
        o.h[3] = (__bf16)((d3 * inv * scv.w + bbv.w) * (float)g.h[3]);
        *(ushort4*)(mytile + rr * D128 + lane * 4) = o.u;
    }

    v16bf a[4];
#pragma unroll
    for (int kk = 0; kk < 4; ++kk) a[kk] = frag_load(mytile, 0, D128, kk * 32, lane);

    for (int nt = 0; nt < 8; ++nt) {
        v8f acc = {};
        acc = gemm_tile128(a, Wt, nt * 16, lane, acc);
        float bb = bso[nt * 16 + nl];
#pragma unroll
        for (int v = 0; v < 8; ++v) {
            size_t r = r0 + v + 8 * hl;
            y[r * D128 + nt * 16 + nl] = acc[v] + bb;
        }
    }
}

// ---------------------------------------------------------------------------
extern "C" void kernel_launch(void* const* d_in, const int* in_sizes, int n_in,
                              void* d_out, int out_size, void* d_ws, size_t ws_size,
                              hipStream_t stream) {
    const float* x    = (const float*)d_in[0];
    const float* msk  = (const float*)d_in[1];
    const float* nsc  = (const float*)d_in[2];
    const float* nbi  = (const float*)d_in[3];
    const float* Wl   = (const float*)d_in[4];
    const float* bl   = (const float*)d_in[5];
    const float* Wr   = (const float*)d_in[6];
    const float* br   = (const float*)d_in[7];
    const float* Wlg  = (const float*)d_in[8];
    const float* blg  = (const float*)d_in[9];
    const float* Wrg  = (const float*)d_in[10];
    const float* brg  = (const float*)d_in[11];
    const float* Wog  = (const float*)d_in[12];
    const float* bog  = (const float*)d_in[13];
    const float* ons  = (const float*)d_in[14];
    const float* onb  = (const float*)d_in[15];
    const float* Wout = (const float*)d_in[16];
    const float* bout = (const float*)d_in[17];
    float* y = (float*)d_out;

    const size_t SZ_BF16 = (size_t)D128 * MROWS * sizeof(bf16_t);  // 64 MB
    char* ws = (char*)d_ws;
    bf16_t* Ag   = (bf16_t*)(ws + 0 * SZ_BF16);
    bf16_t* Bg   = (bf16_t*)(ws + 1 * SZ_BF16);
    bf16_t* gate = (bf16_t*)(ws + 2 * SZ_BF16);
    bf16_t* xn   = (bf16_t*)(ws + 3 * SZ_BF16);          // dead after K2
    float*  tri  = (float*) (ws + 3 * SZ_BF16);          // aliases xn (128 MB)

    // K1: layernorm (8 rows/block)
    k_layernorm<<<MROWS / 8, 256, 0, stream>>>(x, nsc, nbi, xn);

    // K2: fused projections (128 rows/block); 160 KB + 2.5 KB dynamic LDS
    size_t shm2 = 5 * D128 * D128 * sizeof(bf16_t) + 5 * D128 * sizeof(float);
    k_proj<<<MROWS / 128, 256, shm2, stream>>>(xn, msk, Wl, bl, Wr, br,
                                               Wlg, blg, Wrg, brg, Wog, bog,
                                               Ag, Bg, gate);

    // K3: per-channel triangle GEMMs, async double-buffered LDS (40 KB)
    dim3 g3(NTOK / 128, NTOK / 128, D128);
    size_t shm3 = 4 * (size_t)K3_BUFE * sizeof(bf16_t);
    k_triangle<<<g3, 256, shm3, stream>>>(Ag, Bg, tri);

    // K4: layernorm + gate + output projection; 32 KB + 0.5 KB + 32 KB LDS
    size_t shm4 = D128 * D128 * sizeof(bf16_t) + D128 * sizeof(float)
                + 8 * 16 * D128 * sizeof(bf16_t);
    k_outproj<<<MROWS / 128, 256, shm4, stream>>>(tri, gate, ons, onb, Wout, bout, y);
}
